// Net_17051020165436
// MI455X (gfx1250) — matrix operand order repaired and checked by reading, not compiled, not run on hardware
//
#include <hip/hip_runtime.h>
#include <hip/hip_bf16.h>

#define DEV __device__ __forceinline__

typedef _Float16 v16h __attribute__((ext_vector_type(16)));
typedef _Float16 half8 __attribute__((ext_vector_type(8)));
typedef _Float16 half4 __attribute__((ext_vector_type(4)));
typedef float v8f __attribute__((ext_vector_type(8)));

static constexpr int BB = 16;     // batch
static constexpr int N1 = 2048;   // nodes stage 1
static constexpr int N2 = 1024;   // nodes stage 2
static constexpr int DD = 4;      // input dim
static constexpr int HH = 64;     // hidden
static constexpr int KK = 8;      // kNN
#define BIGF 1e30f

DEV float eluf(float x) { return x > 0.f ? x : (__expf(x) - 1.f); }

// monotone float -> uint mapping (order preserving), for atomic max on f32
DEV unsigned enc_f32(float v) {
    unsigned b = __float_as_uint(v);
    return (b & 0x80000000u) ? ~b : (b | 0x80000000u);
}
DEV float dec_f32(unsigned k) {
    unsigned b = (k & 0x80000000u) ? (k ^ 0x80000000u) : ~k;
    return __uint_as_float(b);
}

DEV v8f wmma_f16(v16h a, v16h b, v8f c) {
    return __builtin_amdgcn_wmma_f32_16x16x32_f16(false, a, false, b, (short)0, c, false, false);
}

// CDNA5 async global->LDS copy (per-lane b128), tracked by ASYNCcnt
DEV void async_cp_b128(unsigned lds_off, const void* gaddr) {
    asm volatile("global_load_async_to_lds_b128 %0, %1, off" :: "v"(lds_off), "v"(gaddr) : "memory");
}

// ---------------------------------------------------------------- inputnet
__global__ void k_inputnet(const float* __restrict__ x, const float* __restrict__ norm,
                           const float* __restrict__ w1, const float* __restrict__ b1,
                           const float* __restrict__ w2, const float* __restrict__ b2,
                           float* __restrict__ h, _Float16* __restrict__ hf,
                           float* __restrict__ d2, int total) {
    __shared__ float sw1[DD * 32], sb1[32], sw2[32 * HH], sb2[HH];
    for (int i = threadIdx.x; i < DD * 32; i += blockDim.x) sw1[i] = w1[i];
    for (int i = threadIdx.x; i < 32; i += blockDim.x) sb1[i] = b1[i];
    for (int i = threadIdx.x; i < 32 * HH; i += blockDim.x) sw2[i] = w2[i];
    for (int i = threadIdx.x; i < HH; i += blockDim.x) sb2[i] = b2[i];
    __syncthreads();
    int node = blockIdx.x * blockDim.x + threadIdx.x;
    if (node >= total) return;
    float xn[DD];
#pragma unroll
    for (int d = 0; d < DD; ++d) xn[d] = x[node * DD + d] * norm[d];
    float m1[32];
#pragma unroll
    for (int j = 0; j < 32; ++j) {
        float a = sb1[j];
#pragma unroll
        for (int d = 0; d < DD; ++d) a += xn[d] * sw1[d * 32 + j];
        m1[j] = eluf(a);
    }
    float s = 0.f;
    for (int j = 0; j < HH; ++j) {
        float a = sb2[j];
#pragma unroll
        for (int t = 0; t < 32; ++t) a += m1[t] * sw2[t * HH + j];
        float v = eluf(a);
        h[(size_t)node * HH + j] = v;
        hf[(size_t)node * HH + j] = (_Float16)v;
        s += v * v;
    }
    d2[node] = s;
}

// ---------------------------------------------------------------- fill (scatter-max init)
__global__ void k_fill(unsigned* __restrict__ p, int n) {
    int i = blockIdx.x * blockDim.x + threadIdx.x;
    if (i < n) p[i] = enc_f32(-BIGF);
}

// ---------------------------------------------------------------- fused WMMA kNN top-8
// One wave owns 32 rows; column tiles of 16 are double-buffered into LDS with
// async b128 copies; gram via WMMA; per-row top-8 kept in registers with a
// threshold early-out before the insertion network.
__global__ void __launch_bounds__(32) k_knn(const _Float16* __restrict__ hf,
                                            const float* __restrict__ d2,
                                            int* __restrict__ idx, int N) {
    const int lane = threadIdx.x;
    const int rbPerEvent = N / 32;
    const int b = blockIdx.x / rbPerEvent;
    const int rb = (blockIdx.x % rbPerEvent) * 32;
    const _Float16* he = hf + (size_t)b * N * HH;
    const float* d2e = d2 + (size_t)b * N;
    __shared__ float tile[32 * 16];
    __shared__ __align__(16) _Float16 colbuf[2][16 * HH];   // 2 x 2KB double buffer

    // A fragments: rows rb..rb+31, both 32-wide k-chunks, loaded once.
    const int rA = rb + (lane & 15);
    const int baseA = (lane >> 4) * 8;
    v16h a[2][2];
#pragma unroll
    for (int g = 0; g < 2; ++g) {
        const _Float16* rp = he + (size_t)(rA + g * 16) * HH;
#pragma unroll
        for (int c = 0; c < 2; ++c) {
            union { v16h v; half8 h2[2]; } u;
            u.h2[0] = *(const half8*)(rp + c * 32 + baseA);
            u.h2[1] = *(const half8*)(rp + c * 32 + baseA + 16);
            a[g][c] = u.v;
        }
    }

    const unsigned ldsBase0 = (unsigned)(uintptr_t)(void*)&colbuf[0][0];
    const unsigned ldsBase1 = (unsigned)(uintptr_t)(void*)&colbuf[1][0];
    const unsigned laneByte = (unsigned)lane * 16u;

    // prefetch col tile 0 into buffer 0 (2KB contiguous = 4 async b128 per wave)
    {
        const char* src = (const char*)he;
#pragma unroll
        for (int i = 0; i < 4; ++i)
            async_cp_b128(ldsBase0 + i * 512 + laneByte, src + i * 512 + laneByte);
    }

    const int iRow = rb + lane;
    float td[KK]; int ti[KK];
#pragma unroll
    for (int s = 0; s < KK; ++s) { td[s] = BIGF; ti[s] = 0; }

    const int nB = lane & 15;
    const int sB = (lane >> 4) * 8;

    for (int cb = 0; cb < N; cb += 16) {
        const int cur = (cb >> 4) & 1;
        const unsigned curBase = cur ? ldsBase1 : ldsBase0;
        const unsigned nxtBase = cur ? ldsBase0 : ldsBase1;
        if (cb + 16 < N) {
            const char* src = (const char*)(he + (size_t)(cb + 16) * HH);
#pragma unroll
            for (int i = 0; i < 4; ++i)
                async_cp_b128(nxtBase + i * 512 + laneByte, src + i * 512 + laneByte);
            asm volatile("s_wait_asynccnt 0x4" ::: "memory");   // tile cb landed
        } else {
            asm volatile("s_wait_asynccnt 0x0" ::: "memory");
        }
        float cd2 = d2e[cb + nB];
        // B fragments from staged LDS tile
        const _Float16* cp = colbuf[cur] + nB * HH;
        v16h bf[2];
#pragma unroll
        for (int c = 0; c < 2; ++c) {
            union { v16h v; half8 h2[2]; } u;
            u.h2[0] = *(const half8*)(cp + c * 32 + sB);
            u.h2[1] = *(const half8*)(cp + c * 32 + sB + 16);
            bf[c] = u.v;
        }
        v8f acc0 = {}; v8f acc1 = {};
        acc0 = wmma_f16(a[0][0], bf[0], acc0);
        acc0 = wmma_f16(a[0][1], bf[1], acc0);
        acc1 = wmma_f16(a[1][0], bf[0], acc1);
        acc1 = wmma_f16(a[1][1], bf[1], acc1);
#pragma unroll
        for (int q = 0; q < 8; ++q) {
            int m0 = q + 8 * (lane >> 4);
            tile[m0 * 16 + nB] = cd2 - 2.f * acc0[q];
            tile[(m0 + 16) * 16 + nB] = cd2 - 2.f * acc1[q];
        }
        asm volatile("s_wait_dscnt 0x0" ::: "memory");
#pragma unroll
        for (int n = 0; n < 16; ++n) {
            int j = cb + n;
            float dv = tile[lane * 16 + n];
            if (j == iRow) dv = BIGF;
            if (dv < td[KK - 1]) {          // threshold early-out
                int jj = j;
#pragma unroll
                for (int s = 0; s < KK; ++s) {   // register insertion network
                    bool sw = dv < td[s];
                    float nd = sw ? dv : td[s]; int nj = sw ? jj : ti[s];
                    dv = sw ? td[s] : dv;        jj = sw ? ti[s] : jj;
                    td[s] = nd; ti[s] = nj;
                }
            }
        }
    }
    int* op = idx + ((size_t)b * N + iRow) * KK;
#pragma unroll
    for (int s = 0; s < KK; ++s) op[s] = ti[s];
}

// ---------------------------------------------------------------- WMMA EdgeConv
// 2 waves/block; each wave processes 16-edge tiles: stage [xi, xj-xi] (16x128 f16)
// in LDS A-layout, GEMM1 128->96 (24 WMMA), ELU, GEMM2 96->64 (12 WMMA), ELU,
// scatter-max via monotone-uint atomicMax.
__global__ void __launch_bounds__(64) k_edgeconv(const float* __restrict__ h,
                                                 const int* __restrict__ idx,
                                                 const float* __restrict__ wa, const float* __restrict__ ba_,
                                                 const float* __restrict__ wb, const float* __restrict__ bb_,
                                                 unsigned* __restrict__ out_u, int N) {
    __shared__ _Float16 swa[96 * 128];       // w_a transposed [n][k]
    __shared__ _Float16 swb[64 * 96];        // w_b transposed [n][k]
    __shared__ float sba[96], sbb[64];
    __shared__ __align__(16) _Float16 feat[2][16 * 128];
    __shared__ __align__(16) _Float16 mid[2][16 * 96];
    __shared__ int sdst[2][16];

    for (int i = threadIdx.x; i < 96 * 128; i += blockDim.x) {
        int n = i >> 7, k = i & 127;
        swa[i] = (_Float16)wa[k * 96 + n];
    }
    for (int i = threadIdx.x; i < 64 * 96; i += blockDim.x) {
        int n = i / 96, k = i - n * 96;
        swb[i] = (_Float16)wb[k * 64 + n];
    }
    for (int i = threadIdx.x; i < 96; i += blockDim.x) sba[i] = ba_[i];
    for (int i = threadIdx.x; i < 64; i += blockDim.x) sbb[i] = bb_[i];
    __syncthreads();

    const int wave = threadIdx.x >> 5;
    const int lane = threadIdx.x & 31;
    _Float16* F = feat[wave];
    _Float16* Mi = mid[wave];
    int* DS = sdst[wave];

    const int NK = N * KK;
    const int tilesPerEvent = N;             // 2*N*K/16
    const int totalTiles = tilesPerEvent * BB;
    const int wgid = blockIdx.x * 2 + wave;
    const int wstride = gridDim.x * 2;

    const int r = lane & 15;                 // edge row in tile / A-matrix M
    const int g = lane >> 4;                 // k-half selector
    const int nB = lane & 15;                // B/C column
    const int sB = g * 16;
    const int baseA = g * 16;

    for (int t = wgid; t < totalTiles; t += wstride) {
        int b = t / tilesPerEvent;
        int tt = t - b * tilesPerEvent;
        const float* he = h + (size_t)b * N * HH;
        const int* ie = idx + (size_t)b * N * KK;
        unsigned* oe = out_u + (size_t)b * N * HH;

        // build edge features: e = tt*16 + r
        int e = tt * 16 + r;
        int src, dst;
        if (e < NK) { dst = e >> 3; src = ie[e]; }
        else        { int e2 = e - NK; src = e2 >> 3; dst = ie[e2]; }
        if (g == 0) DS[r] = dst;
        const float4* pd4 = (const float4*)(he + (size_t)dst * HH + g * 32);
        const float4* ps4 = (const float4*)(he + (size_t)src * HH + g * 32);
#pragma unroll
        for (int k4 = 0; k4 < 8; ++k4) {
            float4 xi = pd4[k4], xj = ps4[k4];
            half4 va = { (_Float16)xi.x, (_Float16)xi.y, (_Float16)xi.z, (_Float16)xi.w };
            half4 vd = { (_Float16)(xj.x - xi.x), (_Float16)(xj.y - xi.y),
                         (_Float16)(xj.z - xi.z), (_Float16)(xj.w - xi.w) };
            *(half4*)(F + r * 128 + g * 32 + k4 * 4) = va;
            *(half4*)(F + r * 128 + 64 + g * 32 + k4 * 4) = vd;
        }
        asm volatile("s_wait_dscnt 0x0" ::: "memory");

        // GEMM1: [16x128] x [128x96]
#pragma unroll
        for (int nt = 0; nt < 6; ++nt) {
            v8f acc = {};
#pragma unroll
            for (int c = 0; c < 4; ++c) {
                int kc = c * 32;
                union { v16h v; half8 h2[2]; } ua, ub;
                ua.h2[0] = *(const half8*)(F + r * 128 + kc + baseA);
                ua.h2[1] = *(const half8*)(F + r * 128 + kc + baseA + 8);
                const _Float16* wp = swa + (nt * 16 + nB) * 128 + kc + sB;
                ub.h2[0] = *(const half8*)(wp);
                ub.h2[1] = *(const half8*)(wp + 8);
                acc = wmma_f16(ua.v, ub.v, acc);
            }
            float bias = sba[nt * 16 + nB];
#pragma unroll
            for (int q = 0; q < 8; ++q) {
                int m = q + 8 * g;
                Mi[m * 96 + nt * 16 + nB] = (_Float16)eluf(acc[q] + bias);
            }
        }
        asm volatile("s_wait_dscnt 0x0" ::: "memory");

        // GEMM2: [16x96] x [96x64] + scatter-max
#pragma unroll
        for (int nt = 0; nt < 4; ++nt) {
            v8f acc = {};
#pragma unroll
            for (int c = 0; c < 3; ++c) {
                int kc = c * 32;
                union { v16h v; half8 h2[2]; } ua, ub;
                ua.h2[0] = *(const half8*)(Mi + r * 96 + kc + baseA);
                ua.h2[1] = *(const half8*)(Mi + r * 96 + kc + baseA + 8);
                const _Float16* wp = swb + (nt * 16 + nB) * 96 + kc + sB;
                ub.h2[0] = *(const half8*)(wp);
                ub.h2[1] = *(const half8*)(wp + 8);
                acc = wmma_f16(ua.v, ub.v, acc);
            }
            float bias = sbb[nt * 16 + nB];
#pragma unroll
            for (int q = 0; q < 8; ++q) {
                int m = q + 8 * g;
                float v = eluf(acc[q] + bias);
                atomicMax(&oe[(size_t)DS[m] * HH + nt * 16 + nB], enc_f32(v));
            }
        }
        asm volatile("s_wait_dscnt 0x0" ::: "memory");
    }
}

// ---------------------------------------------------------------- coarsen (pairwise max) + norms
__global__ void k_coarsen(const unsigned* __restrict__ h1u, float* __restrict__ h1c,
                          _Float16* __restrict__ h1cf, float* __restrict__ d2o, int total) {
    int n = blockIdx.x * blockDim.x + threadIdx.x;
    if (n >= total) return;
    const unsigned* p0 = h1u + (size_t)(2 * n) * HH;
    const unsigned* p1 = p0 + HH;
    float s = 0.f;
    for (int d = 0; d < HH; ++d) {
        float v = fmaxf(dec_f32(p0[d]), dec_f32(p1[d]));
        h1c[(size_t)n * HH + d] = v;
        h1cf[(size_t)n * HH + d] = (_Float16)v;
        s += v * v;
    }
    d2o[n] = s;
}

// ---------------------------------------------------------------- global max-pool + head MLP
__global__ void __launch_bounds__(64) k_finalize(const unsigned* __restrict__ h2u,
                                                 const float* __restrict__ wo1, const float* __restrict__ bo1,
                                                 const float* __restrict__ wo2, const float* __restrict__ bo2,
                                                 const float* __restrict__ wo3, const float* __restrict__ bo3,
                                                 float* __restrict__ out, int N) {
    int b = blockIdx.x;
    int t = threadIdx.x;
    __shared__ float gg[64], o1[64], o2[32];
    const unsigned* p = h2u + (size_t)b * N * HH + t;
    float m = -BIGF;
    for (int n = 0; n < N; ++n) m = fmaxf(m, dec_f32(p[(size_t)n * HH]));
    gg[t] = m;
    __syncthreads();
    float a1 = bo1[t];
    for (int k = 0; k < 64; ++k) a1 += gg[k] * wo1[k * 64 + t];
    o1[t] = eluf(a1);
    __syncthreads();
    if (t < 32) {
        float a2 = bo2[t];
        for (int k = 0; k < 64; ++k) a2 += o1[k] * wo2[k * 32 + t];
        o2[t] = eluf(a2);
    }
    __syncthreads();
    if (t < 2) {
        float a3 = bo3[t];
        for (int k = 0; k < 32; ++k) a3 += o2[k] * wo3[k * 2 + t];
        if (t == 0) {
            out[b * 3 + 0] = logf(1.f + __expf(a3));            // softplus
        } else {
            float ang = fminf(fmaxf(a3, -3.14159274f), 3.14159274f);
            out[b * 3 + 1] = __cosf(ang);
            out[b * 3 + 2] = __sinf(ang);
        }
    }
}

// ---------------------------------------------------------------- launch
extern "C" void kernel_launch(void* const* d_in, const int* in_sizes, int n_in,
                              void* d_out, int out_size, void* d_ws, size_t ws_size,
                              hipStream_t stream) {
    const float* x     = (const float*)d_in[0];
    const float* norm  = (const float*)d_in[1];
    const float* w_in1 = (const float*)d_in[2];
    const float* b_in1 = (const float*)d_in[3];
    const float* w_in2 = (const float*)d_in[4];
    const float* b_in2 = (const float*)d_in[5];
    const float* w_c1a = (const float*)d_in[6];
    const float* b_c1a = (const float*)d_in[7];
    const float* w_c1b = (const float*)d_in[8];
    const float* b_c1b = (const float*)d_in[9];
    const float* w_c2a = (const float*)d_in[10];
    const float* b_c2a = (const float*)d_in[11];
    const float* w_c2b = (const float*)d_in[12];
    const float* b_c2b = (const float*)d_in[13];
    const float* w_o1  = (const float*)d_in[14];
    const float* b_o1  = (const float*)d_in[15];
    const float* w_o2  = (const float*)d_in[16];
    const float* b_o2  = (const float*)d_in[17];
    const float* w_o3  = (const float*)d_in[18];
    const float* b_o3  = (const float*)d_in[19];
    float* out = (float*)d_out;

    char* p = (char*)d_ws;
    auto carve = [&](size_t bytes) { void* q = (void*)p; p += (bytes + 255) & ~(size_t)255; return q; };
    float*     h_f32  = (float*)    carve((size_t)BB * N1 * HH * sizeof(float));
    _Float16*  h_f16  = (_Float16*) carve((size_t)BB * N1 * HH * sizeof(_Float16));
    float*     d2a    = (float*)    carve((size_t)BB * N1 * sizeof(float));
    int*       idx1   = (int*)      carve((size_t)BB * N1 * KK * sizeof(int));
    unsigned*  h1_u   = (unsigned*) carve((size_t)BB * N1 * HH * sizeof(unsigned));
    float*     h1c    = (float*)    carve((size_t)BB * N2 * HH * sizeof(float));
    _Float16*  h1cf   = (_Float16*) carve((size_t)BB * N2 * HH * sizeof(_Float16));
    float*     d2b    = (float*)    carve((size_t)BB * N2 * sizeof(float));
    int*       idx2   = (int*)      carve((size_t)BB * N2 * KK * sizeof(int));
    unsigned*  h2_u   = (unsigned*) carve((size_t)BB * N2 * HH * sizeof(unsigned));
    (void)ws_size; (void)n_in; (void)in_sizes; (void)out_size;

    // 1) inputnet
    {
        int total = BB * N1;
        k_inputnet<<<(total + 255) / 256, 256, 0, stream>>>(x, norm, w_in1, b_in1, w_in2, b_in2,
                                                            h_f32, h_f16, d2a, total);
    }
    // 2) init scatter-max buffers
    {
        int n1 = BB * N1 * HH, n2 = BB * N2 * HH;
        k_fill<<<(n1 + 255) / 256, 256, 0, stream>>>(h1_u, n1);
        k_fill<<<(n2 + 255) / 256, 256, 0, stream>>>(h2_u, n2);
    }
    // 3) kNN stage 1 (WMMA + async double-buffered tiles)
    k_knn<<<BB * (N1 / 32), 32, 0, stream>>>(h_f16, d2a, idx1, N1);
    // 4) EdgeConv 1 (WMMA)
    k_edgeconv<<<512, 64, 0, stream>>>(h_f32, idx1, w_c1a, b_c1a, w_c1b, b_c1b, h1_u, N1);
    // 5) coarsen
    {
        int total = BB * N2;
        k_coarsen<<<(total + 255) / 256, 256, 0, stream>>>(h1_u, h1c, h1cf, d2b, total);
    }
    // 6) kNN stage 2
    k_knn<<<BB * (N2 / 32), 32, 0, stream>>>(h1cf, d2b, idx2, N2);
    // 7) EdgeConv 2
    k_edgeconv<<<512, 64, 0, stream>>>(h1c, idx2, w_c2a, b_c2a, w_c2b, b_c2b, h2_u, N2);
    // 8) global max-pool + head
    k_finalize<<<BB, 64, 0, stream>>>(h2_u, w_o1, b_o1, w_o2, b_o2, w_o3, b_o3, out, N2);
}
